// Mamba3Block_62010737819995
// MI455X (gfx1250) — matrix operand order repaired and checked by reading, not compiled, not run on hardware
//
#include <hip/hip_runtime.h>

// ---------------- problem constants ----------------
#define BSZ      4
#define SEQL     4096
#define M_TOT    (BSZ*SEQL)      // 16384 rows
#define DMODEL   1024
#define EEXP     2048            // E
#define DSTATE   128
#define THETA    64
#define ZDIM     4419
#define ZPAD     4480            // multiple of 64 (gemm wave-tile N)
// z row layout: x[0,2048) gate[2048,4096) B[4096,4224) C[4224,4352)
//               dt[4352] Alog[4353] theta[4354,4418) lam[4418]

typedef __attribute__((ext_vector_type(16))) __bf16 bf16x16;
typedef __attribute__((ext_vector_type(8)))  __bf16 bf16x8;
typedef __attribute__((ext_vector_type(8)))  float  f32x8;

#define WMMA_BF16(a, b, c) \
    __builtin_amdgcn_wmma_f32_16x16x32_bf16(false, (a), false, (b), (short)0, (c), false, false)

// ---------------- bf16 WMMA GEMM: C[M,N] = A[M,K] * Bt[N,K]^T ----------------
// one wave -> 16(M) x 64(N) tile (4 accumulators, A fragment reused 4x)
// block of 8 waves stacked in M -> 128(M) x 64(N) macro tile
// software-pipelined: fragments for k+32 are loaded before the WMMAs for k issue
__global__ __launch_bounds__(256) void gemm_bf16_wmma(
    const __bf16* __restrict__ A,    // M x K, row-major
    const __bf16* __restrict__ Bt,   // N x K, row-major (i.e. B transposed)
    float* __restrict__ C,           // M x N, row-major
    int M, int N, int K)
{
    const int lane = threadIdx.x & 31;
    const int wave = threadIdx.x >> 5;                 // 0..7
    const int m0   = blockIdx.y * 128 + wave * 16;
    const int n0   = blockIdx.x * 64;
    const int half = lane >> 4;                        // 0 | 1
    const int l16  = lane & 15;

    // A fragment: row = m0+l16; lanes<16 hold K 0..15, lanes>=16 hold 16..31
    const __bf16* ap = A + (size_t)(m0 + l16) * K + half * 16;
    // B fragments: col = n0+16j+l16; lanes<16 hold K 0..15, lanes>=16 hold 16..31
    const __bf16* bp0 = Bt + (size_t)(n0 +  0 + l16) * K + half * 16;
    const __bf16* bp1 = Bt + (size_t)(n0 + 16 + l16) * K + half * 16;
    const __bf16* bp2 = Bt + (size_t)(n0 + 32 + l16) * K + half * 16;
    const __bf16* bp3 = Bt + (size_t)(n0 + 48 + l16) * K + half * 16;

    f32x8 acc0 = {}, acc1 = {}, acc2 = {}, acc3 = {};

    // preload k = 0
    bf16x16 a, b0, b1, b2, b3;
    *(bf16x8*)&a       = *(const bf16x8*)(ap);
    *((bf16x8*)&a + 1) = *(const bf16x8*)(ap + 8);
    b0 = *(const bf16x16*)(bp0);
    b1 = *(const bf16x16*)(bp1);
    b2 = *(const bf16x16*)(bp2);
    b3 = *(const bf16x16*)(bp3);

    for (int k = 32; k < K; k += 32) {
        // issue next-step loads first so they overlap the WMMAs below
        bf16x16 an, bn0, bn1, bn2, bn3;
        *(bf16x8*)&an       = *(const bf16x8*)(ap + k);
        *((bf16x8*)&an + 1) = *(const bf16x8*)(ap + k + 8);
        bn0 = *(const bf16x16*)(bp0 + k);
        bn1 = *(const bf16x16*)(bp1 + k);
        bn2 = *(const bf16x16*)(bp2 + k);
        bn3 = *(const bf16x16*)(bp3 + k);
        __builtin_prefetch(ap + k + 128, 0, 1);        // A stream, ~4 steps ahead

        acc0 = WMMA_BF16(a, b0, acc0);
        acc1 = WMMA_BF16(a, b1, acc1);
        acc2 = WMMA_BF16(a, b2, acc2);
        acc3 = WMMA_BF16(a, b3, acc3);

        a = an; b0 = bn0; b1 = bn1; b2 = bn2; b3 = bn3;
    }
    acc0 = WMMA_BF16(a, b0, acc0);
    acc1 = WMMA_BF16(a, b1, acc1);
    acc2 = WMMA_BF16(a, b2, acc2);
    acc3 = WMMA_BF16(a, b3, acc3);

    // store: C/D layout: VGPR j -> row m0 + 8*half + j, col = n-tile base + l16
    const int row0 = m0 + half * 8;
    if (row0 + 7 < M) {
        #pragma unroll
        for (int j = 0; j < 8; ++j) {
            float* cr = C + (size_t)(row0 + j) * N + n0 + l16;
            cr[ 0] = acc0[j];
            cr[16] = acc1[j];
            cr[32] = acc2[j];
            cr[48] = acc3[j];
        }
    }
}

// ---------------- converters ----------------
__global__ void cvt_u_bf16(const float* __restrict__ u, __bf16* __restrict__ o, int n) {
    int i = blockIdx.x * 256 + threadIdx.x;
    if (i < n) o[i] = (__bf16)u[i];
}
// WinT[n][k] = W_in[k][n], n in [0,ZPAD), zero-pad n >= ZDIM
__global__ void cvt_winT(const float* __restrict__ Win, __bf16* __restrict__ o) {
    int i = blockIdx.x * 256 + threadIdx.x;               // over ZPAD*DMODEL
    if (i >= ZPAD * DMODEL) return;
    int n = i / DMODEL, k = i - n * DMODEL;
    o[(size_t)n * DMODEL + k] = (n < ZDIM) ? (__bf16)Win[(size_t)k * ZDIM + n] : (__bf16)0.0f;
}
// WoutT[n][k] = W_out[k][n], n in [0,DMODEL), k in [0,EEXP)
__global__ void cvt_woutT(const float* __restrict__ Wout, __bf16* __restrict__ o) {
    int i = blockIdx.x * 256 + threadIdx.x;               // over DMODEL*EEXP
    if (i >= DMODEL * EEXP) return;
    int n = i / EEXP, k = i - n * EEXP;
    o[(size_t)n * EEXP + k] = (__bf16)Wout[(size_t)k * DMODEL + n];
}
__global__ void zero_scalar(float* p) { if (threadIdx.x == 0) p[0] = 0.0f; }

// ---------------- pointwise pass 1 ----------------
// per position: silu(x) in place; rmsnorm(B),rmsnorm(C)+bias; dt/A/lam; dangle; A-sum
__global__ __launch_bounds__(256) void pointwise1(
    float* __restrict__ z,
    const float* __restrict__ wB, const float* __restrict__ wC,
    const float* __restrict__ Bb, const float* __restrict__ Cb,
    const float* __restrict__ dt_bias, const float* __restrict__ A_log,
    float* __restrict__ Bn, float* __restrict__ Cn,
    float* __restrict__ dtv, float* __restrict__ Av, float* __restrict__ lamv,
    float* __restrict__ dangle, float* __restrict__ A_sum)
{
    const int pos = blockIdx.x;
    float* zr = z + (size_t)pos * ZPAD;
    const int t = threadIdx.x;
    __shared__ float sb[128], sc[128];
    __shared__ float rb, rc, dt_s;

    for (int i = t; i < EEXP; i += 256) {                 // silu(x) in place
        float v = zr[i];
        zr[i] = v / (1.0f + __expf(-v));
    }
    if (t < 128) {
        float b = zr[4096 + t], c = zr[4224 + t];
        sb[t] = b * b; sc[t] = c * c;
    }
    __syncthreads();
    for (int s = 64; s > 0; s >>= 1) {
        if (t < s) { sb[t] += sb[t + s]; sc[t] += sc[t + s]; }
        __syncthreads();
    }
    if (t == 0) {
        rb = rsqrtf(sb[0] * (1.0f / 128.0f) + 1e-5f);
        rc = rsqrtf(sc[0] * (1.0f / 128.0f) + 1e-5f);
        float dtr = zr[4352] + dt_bias[0];
        float dt  = (dtr > 20.0f) ? dtr : log1pf(__expf(dtr));
        float ar  = zr[4353] + A_log[0];
        float Ax  = -((ar > 20.0f) ? ar : log1pf(__expf(ar)));
        dtv[pos] = dt; Av[pos] = Ax;
        lamv[pos] = 1.0f / (1.0f + __expf(-zr[4418]));
        dt_s = dt;
        atomicAdd(A_sum, Ax);
    }
    __syncthreads();
    if (t < 128) {
        Bn[(size_t)pos * 128 + t] = zr[4096 + t] * rb * wB[t] + Bb[t];
        Cn[(size_t)pos * 128 + t] = zr[4224 + t] * rc * wC[t] + Cb[t];
    }
    if (t < 64) dangle[(size_t)pos * 64 + t] = dt_s * zr[4354 + t];
}

// ---------------- cumsum of dangle over L (per batch, per theta lane) ----------------
__global__ __launch_bounds__(64) void cumsum_angle(
    const float* __restrict__ dangle, float* __restrict__ cum)
{
    const int b = blockIdx.x, t = threadIdx.x;            // 64 threads
    const float* dp = dangle + (size_t)b * SEQL * 64 + t;
    float*       cp = cum    + (size_t)b * SEQL * 64 + t;
    float acc = 0.0f;
    for (int l = 0; l < SEQL; ++l) {
        acc += dp[(size_t)l * 64];
        cp[(size_t)l * 64] = acc;
    }
}

// ---------------- pointwise pass 2: RoPE, x_trap mean, dBu, dA ----------------
__global__ __launch_bounds__(128) void pointwise2(
    const float* __restrict__ z, const float* __restrict__ Bn, const float* __restrict__ Cn,
    const float* __restrict__ cum, const float* __restrict__ dtv, const float* __restrict__ Av,
    const float* __restrict__ lamv, const float* __restrict__ A_sum,
    float* __restrict__ Br, float* __restrict__ Cr,
    float* __restrict__ dBu, float* __restrict__ dAv)
{
    const int pos = blockIdx.x;
    const int l   = pos & (SEQL - 1);
    const int t   = threadIdx.x;
    __shared__ float brs[128];
    __shared__ float ps[4];
    __shared__ float umean_s;

    const float dt  = dtv[pos];
    const float Ax  = Av[pos];
    const float lam = lamv[pos];

    if (t < 64) {                                          // RoPE on B and C
        float ang = cum[(size_t)pos * 64 + t];
        float c = cosf(ang), s = sinf(ang);
        float b1 = Bn[(size_t)pos * 128 + t], b2 = Bn[(size_t)pos * 128 + 64 + t];
        float c1 = Cn[(size_t)pos * 128 + t], c2 = Cn[(size_t)pos * 128 + 64 + t];
        float br1 = b1 * c - b2 * s, br2 = b1 * s + b2 * c;
        Br[(size_t)pos * 128 + t] = br1;       Br[(size_t)pos * 128 + 64 + t] = br2;
        Cr[(size_t)pos * 128 + t] = c1 * c - c2 * s;
        Cr[(size_t)pos * 128 + 64 + t] = c1 * s + c2 * c;
        brs[t] = br1; brs[64 + t] = br2;
    }

    const float alpha = __expf(dt * Ax);
    const float* xr = z + (size_t)pos * ZPAD;
    const float* xp = z + (size_t)(pos - 1) * ZPAD;        // valid only if l>0
    float sum = 0.0f;
    for (int i = t; i < EEXP; i += 128) {
        float pv = (l == 0) ? 0.0f : xp[i];
        sum += lam * xr[i] + (1.0f - lam) * alpha * pv;
    }
    for (int off = 16; off; off >>= 1) sum += __shfl_down(sum, off, 32);
    if ((t & 31) == 0) ps[t >> 5] = sum;
    __syncthreads();
    if (t == 0) {
        umean_s = (ps[0] + ps[1] + ps[2] + ps[3]) * (1.0f / (float)EEXP);
        dAv[pos] = __expf(dt * (A_sum[0] * (1.0f / (float)M_TOT)));
    }
    __syncthreads();
    dBu[(size_t)pos * 128 + t] = dt * brs[t] * umean_s;
}

// ---------------- SSM scan: state[128] per batch, sequential over L ----------------
__global__ __launch_bounds__(128) void scan_kernel(
    const float* __restrict__ dAv, const float* __restrict__ dBu,
    const float* __restrict__ Cr, float* __restrict__ y_raw)
{
    const int b = blockIdx.x, t = threadIdx.x;             // 128 threads = 4 waves
    __shared__ float ps[4];
    float state = 0.0f;
    const size_t base = (size_t)b * SEQL;
    for (int l = 0; l < SEQL; ++l) {
        const size_t pos = base + l;
        state = dAv[pos] * state + dBu[pos * 128 + t];
        float prod = state * Cr[pos * 128 + t];
        for (int off = 16; off; off >>= 1) prod += __shfl_down(prod, off, 32);
        if ((t & 31) == 0) ps[t >> 5] = prod;
        __syncthreads();
        if (t == 0) y_raw[pos] = ps[0] + ps[1] + ps[2] + ps[3];
        __syncthreads();
    }
}

// ---------------- g2 = silu(gate) * y_raw  (bf16 for GEMM2) ----------------
__global__ void make_g2(const float* __restrict__ z, const float* __restrict__ y_raw,
                        __bf16* __restrict__ g2)
{
    size_t i = (size_t)blockIdx.x * 256 + threadIdx.x;     // over M_TOT*EEXP
    if (i >= (size_t)M_TOT * EEXP) return;
    size_t pos = i / EEXP; int c = (int)(i - pos * EEXP);
    float g = z[pos * ZPAD + 2048 + c];
    float sg = g / (1.0f + __expf(-g));
    g2[i] = (__bf16)(y_raw[pos] * sg);
}

// ---------------- launch ----------------
static inline size_t al256(size_t x) { return (x + 255) & ~(size_t)255; }

extern "C" void kernel_launch(void* const* d_in, const int* in_sizes, int n_in,
                              void* d_out, int out_size, void* d_ws, size_t ws_size,
                              hipStream_t stream) {
    const float* u       = (const float*)d_in[0];
    const float* W_in    = (const float*)d_in[1];
    const float* wB      = (const float*)d_in[2];
    const float* wC      = (const float*)d_in[3];
    const float* B_bias  = (const float*)d_in[4];
    const float* C_bias  = (const float*)d_in[5];
    const float* dt_bias = (const float*)d_in[6];
    const float* A_log   = (const float*)d_in[7];
    const float* W_out   = (const float*)d_in[8];
    float* out = (float*)d_out;

    char* w = (char*)d_ws; size_t o = 0;
    auto alloc = [&](size_t bytes) { void* p = w + o; o += al256(bytes); return p; };
    __bf16* u_bf   = (__bf16*)alloc((size_t)M_TOT * DMODEL * 2);
    __bf16* winT   = (__bf16*)alloc((size_t)ZPAD * DMODEL * 2);
    __bf16* woutT  = (__bf16*)alloc((size_t)DMODEL * EEXP * 2);
    float*  z      = (float*) alloc((size_t)M_TOT * ZPAD * 4);
    float*  Bn     = (float*) alloc((size_t)M_TOT * 128 * 4);
    float*  Cn     = (float*) alloc((size_t)M_TOT * 128 * 4);
    float*  Br     = (float*) alloc((size_t)M_TOT * 128 * 4);
    float*  Cr     = (float*) alloc((size_t)M_TOT * 128 * 4);
    float*  dBu    = (float*) alloc((size_t)M_TOT * 128 * 4);
    float*  dangle = (float*) alloc((size_t)M_TOT * 64 * 4);
    float*  cum    = (float*) alloc((size_t)M_TOT * 64 * 4);
    float*  dtv    = (float*) alloc((size_t)M_TOT * 4);
    float*  Av     = (float*) alloc((size_t)M_TOT * 4);
    float*  lamv   = (float*) alloc((size_t)M_TOT * 4);
    float*  dAv    = (float*) alloc((size_t)M_TOT * 4);
    float*  y_raw  = (float*) alloc((size_t)M_TOT * 4);
    float*  A_sum  = (float*) alloc(256);
    __bf16* g2     = (__bf16*)alloc((size_t)M_TOT * EEXP * 2);

    zero_scalar<<<1, 32, 0, stream>>>(A_sum);

    { int n = M_TOT * DMODEL;
      cvt_u_bf16<<<(n + 255) / 256, 256, 0, stream>>>(u, u_bf, n); }
    cvt_winT <<<(ZPAD * DMODEL + 255) / 256, 256, 0, stream>>>(W_in, winT);
    cvt_woutT<<<(DMODEL * EEXP + 255) / 256, 256, 0, stream>>>(W_out, woutT);

    // GEMM1: z[M_TOT, ZPAD] = u @ W_in   (grid: N/64 x M/128)
    gemm_bf16_wmma<<<dim3(ZPAD / 64, M_TOT / 128), 256, 0, stream>>>(
        u_bf, winT, z, M_TOT, ZPAD, DMODEL);

    pointwise1<<<M_TOT, 256, 0, stream>>>(z, wB, wC, B_bias, C_bias, dt_bias, A_log,
                                          Bn, Cn, dtv, Av, lamv, dangle, A_sum);

    cumsum_angle<<<BSZ, 64, 0, stream>>>(dangle, cum);

    pointwise2<<<M_TOT, 128, 0, stream>>>(z, Bn, Cn, cum, dtv, Av, lamv, A_sum,
                                          Br, Cr, dBu, dAv);

    scan_kernel<<<BSZ, 128, 0, stream>>>(dAv, dBu, Cr, y_raw);

    { size_t n = (size_t)M_TOT * EEXP;
      make_g2<<<(unsigned)((n + 255) / 256), 256, 0, stream>>>(z, y_raw, g2); }

    // GEMM2: out[M_TOT, DMODEL] = g2 @ W_out   (grid: N/64 x M/128)
    gemm_bf16_wmma<<<dim3(DMODEL / 64, M_TOT / 128), 256, 0, stream>>>(
        g2, woutT, out, M_TOT, DMODEL, EEXP);
}
